// BERT_61873298866553
// MI455X (gfx1250) — matrix operand order repaired and checked by reading, not compiled, not run on hardware
//
#include <hip/hip_runtime.h>

typedef __bf16 bf16;
typedef __attribute__((ext_vector_type(16))) __bf16 v16bf;
typedef __attribute__((ext_vector_type(8)))  __bf16 v8bf;
typedef __attribute__((ext_vector_type(8)))  float   v8f;
typedef unsigned int u32x4 __attribute__((ext_vector_type(4)));
typedef int          i32x4 __attribute__((ext_vector_type(4)));
typedef int          i32x8 __attribute__((ext_vector_type(8)));

// Problem constants (match reference)
#define BB   2
#define SS   1024
#define INF_ 64
#define DD   1024
#define HH   16
#define NLL  6
#define DFF  4096
#define DKK  64
#define DRR  256
#define MM   (BB*SS)   // 2048 tokens

// ---------------------------------------------------------------------------
// Tensor Data Mover: async 2D tile DMA  global -> LDS  (D# per ISA 08 §8.3/8.4)
//   tile_x elements contiguous per line (data_size=2B), tile_y lines,
//   line stride = stride_el elements.  LDS receives tile row-major packed.
// Issue from ONE wave only; completion via TENSORcnt.
// ---------------------------------------------------------------------------
__device__ __forceinline__
void tdm_load_tile_2d(unsigned lds_byte_addr, const void* gtile,
                      unsigned tile_x, unsigned tile_y,
                      unsigned long long stride_el)
{
    union { unsigned long long q[2]; u32x4 v; } g0;
    union { unsigned long long q[4]; i32x8 v; } g1;
    // group0: count=1 | lds_addr<<32 ; global byte addr (57b) | type=2 (bits 127:126 = 10b)
    g0.q[0] = 1ull | ((unsigned long long)lds_byte_addr << 32);
    g0.q[1] = (((unsigned long long)(size_t)gtile) & 0x01FFFFFFFFFFFFFFull)
            | 0x8000000000000000ull;
    const unsigned long long td0 = tile_x;   // tensor_dim0 == tile_dim0 (tile in-bounds)
    const unsigned long long td1 = tile_y;   // tensor_dim1 == tile_dim1
    // group1 bit packing (bits relative to 256-bit group):
    //  [17:16] data_size=1 (2 bytes); [79:48] tensor_dim0; [111:80] tensor_dim1;
    //  [127:112] tile_dim0; [143:128] tile_dim1; [207:160] tensor_dim0_stride
    g1.q[0] = (1ull << 16) | (td0 << 48);
    g1.q[1] = (td0 >> 16) | (td1 << 16) | ((unsigned long long)tile_x << 48);
    g1.q[2] = (unsigned long long)tile_y | ((stride_el & 0xFFFFFFFFull) << 32);
    g1.q[3] = (stride_el >> 32) & 0xFFFFull;
    const i32x4 z4 = {0, 0, 0, 0};
    const i32x8 z8 = {0, 0, 0, 0, 0, 0, 0, 0};
    __builtin_amdgcn_tensor_load_to_lds(g0.v, g1.v, z4, z4, z8, 0);
}

// ---------------------------------------------------------------------------
// Generic bf16 WMMA GEMM:  C = act(alpha * A @ B + bias) [+ residual]
//   A: [M,K] bf16 row-major, row stride lda   (per-z offset z1*sA1 + z2*sA2)
//   B: TRANSB=0 -> [K,N] bf16 row-major (ldb = row stride)
//      TRANSB=1 -> [N,K] bf16 row-major (ldb = row stride)  (i.e. B^T)
//   C: [M,N] row stride ldc; OUTF32 ? float : bf16
// 128x128 block tile, BK=32, 256 threads = 8 waves, wave tile 64x32.
// A tile (and B tile when TRANSB=1) staged by the Tensor Data Mover into
// DOUBLE-BUFFERED LDS: the TDM transfer of tile i+1 overlaps the WMMAs of
// tile i (one workgroup barrier per k-step).
// ---------------------------------------------------------------------------
template<int ACT, int OUTF32, int RES, int TRANSB>
__global__ __launch_bounds__(256)
void gemm_wmma(const bf16* __restrict__ A, int lda, long long sA1, long long sA2,
               const bf16* __restrict__ B, int ldb, long long sB1, long long sB2,
               const float* __restrict__ bias,
               void* __restrict__ Cout, int ldc, long long sC1, long long sC2,
               const float* __restrict__ resid,
               int M, int N, int K, int zdiv, float alpha)
{
    // [buf p][A tile 4096 | B tile 4096] bf16  -> 2 x 16 KB
    __shared__ __align__(16) bf16 smem[2 * 2 * 128 * 32];
    const unsigned lds_base = (unsigned)(unsigned long long)(const void*)smem;

    const int tid  = threadIdx.x;
    const int lane = tid & 31;
    const int wave = tid >> 5;
    const int wm   = (wave >> 2) * 64;   // wave M offset: 0 or 64
    const int wn   = (wave & 3) * 32;    // wave N offset: 0,32,64,96

    const int m0 = blockIdx.y * 128;
    const int n0 = blockIdx.x * 128;
    const int z  = blockIdx.z;
    const int z1 = z / zdiv;
    const int z2 = z % zdiv;

    const bf16* Ab = A + (size_t)z1 * sA1 + (size_t)z2 * sA2;
    const bf16* Bb = B + (size_t)z1 * sB1 + (size_t)z2 * sB2;
    const size_t coff = (size_t)z1 * sC1 + (size_t)z2 * sC2;

    v8f acc[4][2];
    const v8f vzero = {0.f,0.f,0.f,0.f,0.f,0.f,0.f,0.f};
#pragma unroll
    for (int mt = 0; mt < 4; ++mt)
#pragma unroll
        for (int nt = 0; nt < 2; ++nt) acc[mt][nt] = vzero;

    const v8bf bzero = {(bf16)0.f,(bf16)0.f,(bf16)0.f,(bf16)0.f,
                        (bf16)0.f,(bf16)0.f,(bf16)0.f,(bf16)0.f};

    // Stage k-tile `kks` into LDS buffer p (TDM for A, and for B when TRANSB;
    // manual transpose staging for row-major [K,N] weights otherwise).
    auto stage_tiles = [&](int p, int kks) {
        if (tid < 32) {
            tdm_load_tile_2d(lds_base + (unsigned)p * 16384u,
                             Ab + (size_t)m0 * lda + kks,
                             32, 128, (unsigned long long)lda);
            if (TRANSB)   // B^T rows are N: same 2D shape -> straight TDM tile
                tdm_load_tile_2d(lds_base + (unsigned)p * 16384u + 8192u,
                                 Bb + (size_t)n0 * ldb + kks,
                                 32, 128, (unsigned long long)ldb);
        }
        if (!TRANSB) {
            bf16* Bsp = smem + p * 8192 + 4096;
#pragma unroll
            for (int i = 0; i < 2; ++i) {
                int chunk = tid + i * 256;      // 512 chunks x 8 elems = 4096
                int kl    = chunk >> 4;         // 0..31
                int nc    = (chunk & 15) << 3;  // 0..120 step 8
                int gn    = n0 + nc;
                v8bf vv   = bzero;
                if (gn < N)                     // N is a multiple of 8 everywhere
                    vv = *(const v8bf*)(Bb + (size_t)(kks + kl) * ldb + gn);
#pragma unroll
                for (int j = 0; j < 8; ++j)
                    Bsp[(nc + j) * 32 + kl] = vv[j];
            }
        }
    };

    // ---- prologue: stage first tile into buffer 0 -------------------------
    stage_tiles(0, 0);

    int p = 0;
    for (int kk = 0; kk < K; kk += 32) {
        if (tid < 32) __builtin_amdgcn_s_wait_tensorcnt(0);
        __syncthreads();   // buffer p ready; all reads of buffer p^1 retired

        // ---- kick off the NEXT tile's DMA while we compute this one -------
        if (kk + 32 < K) stage_tiles(p ^ 1, kk + 32);

        const bf16* Asp = smem + p * 8192;
        const bf16* Bsp = Asp + 4096;

        // ---- fragments per ISA 16-bit A (16x32) / B (32x16) VGPR layouts --
        union UA { v16bf v; v8bf h[2]; };
        UA afr[4], bfr[2];
        const int arow  = lane & 15;
        const int akoff = (lane < 16) ? 0 : 8;   // element offset
#pragma unroll
        for (int mt = 0; mt < 4; ++mt) {
            const bf16* q = &Asp[(wm + mt * 16 + arow) * 32 + akoff];
            afr[mt].h[0] = *(const v8bf*)(q);        // K {koff..koff+7}
            afr[mt].h[1] = *(const v8bf*)(q + 16);   // K {koff+16..koff+23}
        }
        const int bcol  = lane & 15;
    const int bkoff = (lane < 16) ? 0 : 8;
#pragma unroll
        for (int nt = 0; nt < 2; ++nt) {
            const bf16* q = &Bsp[(wn + nt * 16 + bcol) * 32 + bkoff];
            bfr[nt].h[0] = *(const v8bf*)(q);        // K {koff..koff+7}
            bfr[nt].h[1] = *(const v8bf*)(q + 16);   // K {koff+16..koff+23}
        }

        // ---- 8x v_wmma_f32_16x16x32_bf16 ----------------------------------
#pragma unroll
        for (int mt = 0; mt < 4; ++mt)
#pragma unroll
            for (int nt = 0; nt < 2; ++nt)
                acc[mt][nt] = __builtin_amdgcn_wmma_f32_16x16x32_bf16(
                    false, afr[mt].v, false, bfr[nt].v,
                    (short)0, acc[mt][nt], false, false);
        p ^= 1;
    }

    // ---- epilogue: C/D layout: VGPR r -> row r (+8 for lanes>=16) ---------
    const int crow0 = m0 + wm + ((lane < 16) ? 0 : 8);
    const int ccol0 = n0 + wn + (lane & 15);
#pragma unroll
    for (int nt = 0; nt < 2; ++nt) {
        int col = ccol0 + nt * 16;
        if (col >= N) continue;
        float bv = bias ? bias[col] : 0.f;
#pragma unroll
        for (int mt = 0; mt < 4; ++mt) {
#pragma unroll
            for (int r = 0; r < 8; ++r) {
                int row = crow0 + mt * 16 + r;
                if (row >= M) continue;
                float v = acc[mt][nt][r] * alpha + bv;
                if (ACT == 1) v = v > 0.f ? v : 0.f;                       // relu
                else if (ACT == 2) v = 0.5f * v * (1.f + erff(v * 0.70710678118f)); // exact gelu
                size_t idx = coff + (size_t)row * ldc + col;
                if (RES) v += resid[idx];
                if (OUTF32) ((float*)Cout)[idx] = v;
                else        ((bf16*)Cout)[idx] = (bf16)v;
            }
        }
    }
}

// ---------------------------------------------------------------------------
// f32 -> bf16 conversion
// ---------------------------------------------------------------------------
__global__ __launch_bounds__(256)
void cvt_kernel(const float* __restrict__ src, bf16* __restrict__ dst, long long n)
{
    long long i = (long long)blockIdx.x * 256 + threadIdx.x;
    if (i < n) dst[i] = (bf16)src[i];
}

// ---------------------------------------------------------------------------
// LayerNorm: f32 in -> bf16 out, one row (length d) per block
// ---------------------------------------------------------------------------
__global__ __launch_bounds__(256)
void layernorm_kernel(const float* __restrict__ x, const float* __restrict__ g,
                      const float* __restrict__ b, bf16* __restrict__ out, int d)
{
    __shared__ float red[256];
    const int row = blockIdx.x;
    const float* xr = x + (size_t)row * d;
    float s = 0.f;
    for (int i = threadIdx.x; i < d; i += 256) s += xr[i];
    red[threadIdx.x] = s; __syncthreads();
    for (int off = 128; off > 0; off >>= 1) {
        if (threadIdx.x < off) red[threadIdx.x] += red[threadIdx.x + off];
        __syncthreads();
    }
    const float mu = red[0] / d;
    __syncthreads();
    float vs = 0.f;
    for (int i = threadIdx.x; i < d; i += 256) { float dd = xr[i] - mu; vs += dd * dd; }
    red[threadIdx.x] = vs; __syncthreads();
    for (int off = 128; off > 0; off >>= 1) {
        if (threadIdx.x < off) red[threadIdx.x] += red[threadIdx.x + off];
        __syncthreads();
    }
    const float rstd = rsqrtf(red[0] / d + 1e-5f);
    bf16* o = out + (size_t)row * d;
    for (int i = threadIdx.x; i < d; i += 256)
        o[i] = (bf16)((xr[i] - mu) * rstd * g[i] + b[i]);
}

// ---------------------------------------------------------------------------
// Masked softmax, in-place on bf16 rows of length S.  blockIdx.x = b*H*S row.
// ---------------------------------------------------------------------------
__global__ __launch_bounds__(256)
void softmax_kernel(bf16* __restrict__ P, const int* __restrict__ mask)
{
    __shared__ float red[256];
    const int r  = blockIdx.x;
    const int q  = r % SS;
    const int bi = r / (HH * SS);
    bf16* row = P + (size_t)r * SS;
    const int* mrow = mask + (size_t)bi * SS * SS + (size_t)q * SS;

    float mx = -3.0e38f;
    for (int i = threadIdx.x; i < SS; i += 256) {
        float v = (mrow[i] == 0) ? -1.0e9f : (float)row[i];
        mx = fmaxf(mx, v);
    }
    red[threadIdx.x] = mx; __syncthreads();
    for (int off = 128; off > 0; off >>= 1) {
        if (threadIdx.x < off) red[threadIdx.x] = fmaxf(red[threadIdx.x], red[threadIdx.x + off]);
        __syncthreads();
    }
    mx = red[0]; __syncthreads();

    float sum = 0.f;
    for (int i = threadIdx.x; i < SS; i += 256) {
        float v = (mrow[i] == 0) ? -1.0e9f : (float)row[i];
        float e = __expf(v - mx);
        row[i] = (bf16)e;
        sum += e;
    }
    red[threadIdx.x] = sum; __syncthreads();
    for (int off = 128; off > 0; off >>= 1) {
        if (threadIdx.x < off) red[threadIdx.x] += red[threadIdx.x + off];
        __syncthreads();
    }
    const float inv = 1.f / red[0];
    for (int i = threadIdx.x; i < SS; i += 256)
        row[i] = (bf16)((float)row[i] * inv);
}

// ---------------------------------------------------------------------------
// Add sinusoidal positional encoding to f32 residual stream x[b][s][d]
// ---------------------------------------------------------------------------
__global__ __launch_bounds__(256)
void pe_add_kernel(float* __restrict__ x, long long total)
{
    long long idx = (long long)blockIdx.x * 256 + threadIdx.x;
    if (idx >= total) return;
    int d = (int)(idx % DD);
    int s = (int)((idx / DD) % SS);
    int i = d & ~1;
    // 10000^{-i/D} = exp(-i/D * ln(10000))
    float ang = (float)s * __expf(-(float)i * (9.210340372f / (float)DD));
    x[idx] += (d & 1) ? __cosf(ang) : __sinf(ang);
}

// ---------------------------------------------------------------------------
// Final head: out[m] = dot(a[m, 0:K] (bf16), w (f32)) + b0   (N = 1)
// ---------------------------------------------------------------------------
__global__ __launch_bounds__(256)
void head2_kernel(const bf16* __restrict__ a, const float* __restrict__ w,
                  const float* __restrict__ b, float* __restrict__ out, int M, int K)
{
    int m = blockIdx.x * 256 + threadIdx.x;
    if (m >= M) return;
    float s = b[0];
    const bf16* ar = a + (size_t)m * K;
    for (int k = 0; k < K; ++k) s += (float)ar[k] * w[k];
    out[m] = s;
}

// ---------------------------------------------------------------------------
extern "C" void kernel_launch(void* const* d_in, const int* in_sizes, int n_in,
                              void* d_out, int out_size, void* d_ws, size_t ws_size,
                              hipStream_t stream)
{
    (void)in_sizes; (void)n_in; (void)out_size; (void)ws_size;

    const float* src   = (const float*)d_in[0];
    const int*   mask  = (const int*)  d_in[1];
    const float* Wfc1  = (const float*)d_in[2];
    const float* bfc1  = (const float*)d_in[3];
    const float* Wfc2  = (const float*)d_in[4];
    const float* bfc2  = (const float*)d_in[5];
    const float* Wfc3  = (const float*)d_in[6];
    const float* bfc3  = (const float*)d_in[7];
    const float* Wq    = (const float*)d_in[8];
    const float* Wk    = (const float*)d_in[9];
    const float* Wv    = (const float*)d_in[10];
    const float* Wo    = (const float*)d_in[11];
    const float* bo    = (const float*)d_in[12];
    const float* W1    = (const float*)d_in[13];
    const float* b1    = (const float*)d_in[14];
    const float* W2    = (const float*)d_in[15];
    const float* b2    = (const float*)d_in[16];
    const float* ln1_g = (const float*)d_in[17];
    const float* ln1_b = (const float*)d_in[18];
    const float* ln2_g = (const float*)d_in[19];
    const float* ln2_b = (const float*)d_in[20];
    const float* lnf_g = (const float*)d_in[21];
    const float* lnf_b = (const float*)d_in[22];
    const float* Wout1 = (const float*)d_in[23];
    const float* bout1 = (const float*)d_in[24];
    const float* Wout2 = (const float*)d_in[25];
    const float* bout2 = (const float*)d_in[26];
    float* out = (float*)d_out;

    // ---- workspace carve-out ---------------------------------------------
    char* ws = (char*)d_ws;
    size_t off = 0;
    auto alloc = [&](size_t bytes) -> char* {
        off = (off + 255) & ~(size_t)255;
        char* p = ws + off;
        off += bytes;
        return p;
    };
    bf16* wfc1b = (bf16*)alloc((size_t)INF_ * 3072 * 2);
    bf16* wfc2b = (bf16*)alloc((size_t)3072 * 3072 * 2);
    bf16* wfc3b = (bf16*)alloc((size_t)3072 * DD * 2);
    bf16* wqb   = (bf16*)alloc((size_t)NLL * DD * DD * 2);
    bf16* wkb   = (bf16*)alloc((size_t)NLL * DD * DD * 2);
    bf16* wvb   = (bf16*)alloc((size_t)NLL * DD * DD * 2);
    bf16* wob   = (bf16*)alloc((size_t)NLL * DD * DD * 2);
    bf16* w1b   = (bf16*)alloc((size_t)NLL * DD * DFF * 2);
    bf16* w2b   = (bf16*)alloc((size_t)NLL * DFF * DD * 2);
    bf16* wo1b  = (bf16*)alloc((size_t)DD * DRR * 2);
    bf16* srcb  = (bf16*)alloc((size_t)MM * INF_ * 2);
    bf16* a1    = (bf16*)alloc((size_t)MM * 3072 * 2);
    bf16* a2    = (bf16*)alloc((size_t)MM * 3072 * 2);
    float* xf   = (float*)alloc((size_t)MM * DD * 4);
    bf16* x2    = (bf16*)alloc((size_t)MM * DD * 2);
    bf16* qb    = (bf16*)alloc((size_t)MM * DD * 2);
    bf16* kb    = (bf16*)alloc((size_t)MM * DD * 2);
    bf16* vb    = (bf16*)alloc((size_t)MM * DD * 2);
    bf16* ob    = (bf16*)alloc((size_t)MM * DD * 2);
    bf16* hb    = (bf16*)alloc((size_t)MM * DFF * 2);
    bf16* sc    = (bf16*)alloc((size_t)BB * HH * SS * SS * 2);
    bf16* h1    = (bf16*)alloc((size_t)MM * DRR * 2);

    auto cvt = [&](const float* s, bf16* d, long long n) {
        cvt_kernel<<<(unsigned)((n + 255) / 256), 256, 0, stream>>>(s, d, n);
    };

    // ---- convert weights + input to bf16 ---------------------------------
    cvt(Wfc1, wfc1b, (long long)INF_ * 3072);
    cvt(Wfc2, wfc2b, (long long)3072 * 3072);
    cvt(Wfc3, wfc3b, (long long)3072 * DD);
    cvt(Wq,   wqb,   (long long)NLL * DD * DD);
    cvt(Wk,   wkb,   (long long)NLL * DD * DD);
    cvt(Wv,   wvb,   (long long)NLL * DD * DD);
    cvt(Wo,   wob,   (long long)NLL * DD * DD);
    cvt(W1,   w1b,   (long long)NLL * DD * DFF);
    cvt(W2,   w2b,   (long long)NLL * DFF * DD);
    cvt(Wout1, wo1b, (long long)DD * DRR);
    cvt(src,  srcb,  (long long)MM * INF_);

    const long long Z0 = 0;
    // ---- input MLP stack -------------------------------------------------
    {   // a1 = relu(src @ Wfc1 + bfc1)
        dim3 g(3072 / 128, MM / 128, 1);
        gemm_wmma<1,0,0,0><<<g,256,0,stream>>>(srcb, INF_, Z0,Z0, wfc1b, 3072, Z0,Z0,
            bfc1, (void*)a1, 3072, Z0,Z0, nullptr, MM, 3072, INF_, 1, 1.f);
    }
    {   // a2 = relu(a1 @ Wfc2 + bfc2)
        dim3 g(3072 / 128, MM / 128, 1);
        gemm_wmma<1,0,0,0><<<g,256,0,stream>>>(a1, 3072, Z0,Z0, wfc2b, 3072, Z0,Z0,
            bfc2, (void*)a2, 3072, Z0,Z0, nullptr, MM, 3072, 3072, 1, 1.f);
    }
    {   // xf = a2 @ Wfc3 + bfc3   (f32 residual stream)
        dim3 g(DD / 128, MM / 128, 1);
        gemm_wmma<0,1,0,0><<<g,256,0,stream>>>(a2, 3072, Z0,Z0, wfc3b, DD, Z0,Z0,
            bfc3, (void*)xf, DD, Z0,Z0, nullptr, MM, DD, 3072, 1, 1.f);
    }
    // xf += positional encoding
    pe_add_kernel<<<(unsigned)(((long long)MM * DD + 255) / 256), 256, 0, stream>>>(
        xf, (long long)MM * DD);

    const long long sBH_A = (long long)SS * DD;      // per-batch stride in q/k/v/o
    const long long sH_A  = (long long)DKK;          // per-head stride in q/k/v/o
    const long long sB_P  = (long long)HH * SS * SS; // per-batch stride in scores
    const long long sH_P  = (long long)SS * SS;      // per-head stride in scores

    for (int l = 0; l < NLL; ++l) {
        const bf16* wq = wqb + (size_t)l * DD * DD;
        const bf16* wk = wkb + (size_t)l * DD * DD;
        const bf16* wv = wvb + (size_t)l * DD * DD;
        const bf16* wo = wob + (size_t)l * DD * DD;
        const bf16* w1 = w1b + (size_t)l * DD * DFF;
        const bf16* w2 = w2b + (size_t)l * DFF * DD;

        // x2 = LN1(xf)
        layernorm_kernel<<<MM, 256, 0, stream>>>(xf, ln1_g + l * DD, ln1_b + l * DD, x2, DD);

        dim3 gq(DD / 128, MM / 128, 1);
        gemm_wmma<0,0,0,0><<<gq,256,0,stream>>>(x2, DD, Z0,Z0, wq, DD, Z0,Z0,
            nullptr, (void*)qb, DD, Z0,Z0, nullptr, MM, DD, DD, 1, 1.f);
        gemm_wmma<0,0,0,0><<<gq,256,0,stream>>>(x2, DD, Z0,Z0, wk, DD, Z0,Z0,
            nullptr, (void*)kb, DD, Z0,Z0, nullptr, MM, DD, DD, 1, 1.f);
        gemm_wmma<0,0,0,0><<<gq,256,0,stream>>>(x2, DD, Z0,Z0, wv, DD, Z0,Z0,
            nullptr, (void*)vb, DD, Z0,Z0, nullptr, MM, DD, DD, 1, 1.f);

        {   // scores[b,h] = 0.125 * Q_bh @ K_bh^T     (batched, TRANSB via TDM)
            dim3 g(SS / 128, SS / 128, BB * HH);
            gemm_wmma<0,0,0,1><<<g,256,0,stream>>>(
                qb, DD, sBH_A, sH_A,  kb, DD, sBH_A, sH_A,
                nullptr, (void*)sc, SS, sB_P, sH_P, nullptr,
                SS, SS, DKK, HH, 0.125f);
        }
        // softmax over key axis (with mask), in place
        softmax_kernel<<<BB * HH * SS, 256, 0, stream>>>(sc, mask);

        {   // o[b,h] = P_bh @ V_bh   (N = 64 per head)
            dim3 g(1, SS / 128, BB * HH);
            gemm_wmma<0,0,0,0><<<g,256,0,stream>>>(
                sc, SS, sB_P, sH_P,  vb, DD, sBH_A, sH_A,
                nullptr, (void*)ob, DD, sBH_A, sH_A, nullptr,
                SS, DKK, SS, HH, 1.f);
        }
        {   // xf = xf + o @ Wo + bo
            dim3 g(DD / 128, MM / 128, 1);
            gemm_wmma<0,1,1,0><<<g,256,0,stream>>>(ob, DD, Z0,Z0, wo, DD, Z0,Z0,
                bo + l * DD, (void*)xf, DD, Z0,Z0, xf, MM, DD, DD, 1, 1.f);
        }
        // x2 = LN2(xf)
        layernorm_kernel<<<MM, 256, 0, stream>>>(xf, ln2_g + l * DD, ln2_b + l * DD, x2, DD);
        {   // hb = gelu(x2 @ W1 + b1)
            dim3 g(DFF / 128, MM / 128, 1);
            gemm_wmma<2,0,0,0><<<g,256,0,stream>>>(x2, DD, Z0,Z0, w1, DFF, Z0,Z0,
                b1 + l * DFF, (void*)hb, DFF, Z0,Z0, nullptr, MM, DFF, DD, 1, 1.f);
        }
        {   // xf = xf + hb @ W2 + b2
            dim3 g(DD / 128, MM / 128, 1);
            gemm_wmma<0,1,1,0><<<g,256,0,stream>>>(hb, DFF, Z0,Z0, w2, DD, Z0,Z0,
                b2 + l * DD, (void*)xf, DD, Z0,Z0, xf, MM, DD, DFF, 1, 1.f);
        }
    }

    // final LN -> x2
    layernorm_kernel<<<MM, 256, 0, stream>>>(xf, lnf_g, lnf_b, x2, DD);
    {   // h1 = x2 @ Wout1 + bout1
        dim3 g(DRR / 128, MM / 128, 1);
        gemm_wmma<0,0,0,0><<<g,256,0,stream>>>(x2, DD, Z0,Z0, wo1b, DRR, Z0,Z0,
            bout1, (void*)h1, DRR, Z0,Z0, nullptr, MM, DRR, DD, 1, 1.f);
    }
    // out = h1 @ Wout2 + bout2   (N = 1 -> scalar dot per token)
    head2_kernel<<<MM / 256, 256, 0, stream>>>(h1, Wout2, bout2, out, MM, DRR);
}
